// SparseTransformer_83434034692208
// MI455X (gfx1250) — matrix operand order repaired and checked by reading, not compiled, not run on hardware
//
#include <hip/hip_runtime.h>

typedef _Float16 half4  __attribute__((ext_vector_type(4)));
typedef _Float16 half8  __attribute__((ext_vector_type(8)));
typedef _Float16 half16 __attribute__((ext_vector_type(16)));
typedef float    float4v __attribute__((ext_vector_type(4)));
typedef float    float8  __attribute__((ext_vector_type(8)));

// Problem dimensions (fixed by the reference)
constexpr int BB = 4096;    // batch
constexpr int NN = 1024;    // model dim n
constexpr int MM = 16384;   // memory rows m
constexpr int AA = 256;     // attention dim a

// ---------------- CDNA5 async global->LDS copy (ASYNCcnt) ----------------
__device__ __forceinline__ void async_copy_b128(uint32_t ldsAddr, const _Float16* gaddr) {
  // per-lane 16B DMA into LDS; tracked by ASYNCcnt, no VGPR data round-trip
  asm volatile("global_load_async_to_lds_b128 %0, %1, off"
               :: "v"(ldsAddr), "v"(gaddr)
               : "memory");
}
__device__ __forceinline__ void wait_async_all() {
  asm volatile("s_wait_asynccnt 0" ::: "memory");
}

// ---------------- block reductions (wave32) ----------------
__device__ __forceinline__ float blockReduceSum(float v, float* sh) {
#pragma unroll
  for (int o = 16; o > 0; o >>= 1) v += __shfl_xor(v, o, 32);
  const int w  = threadIdx.x >> 5;
  const int nw = blockDim.x >> 5;
  __syncthreads();
  if ((threadIdx.x & 31) == 0) sh[w] = v;
  __syncthreads();
  if (threadIdx.x < 32) {
    float t = (threadIdx.x < nw) ? sh[threadIdx.x] : 0.0f;
#pragma unroll
    for (int o = 16; o > 0; o >>= 1) t += __shfl_xor(t, o, 32);
    if (threadIdx.x == 0) sh[0] = t;
  }
  __syncthreads();
  return sh[0];
}

__device__ __forceinline__ float blockReduceMax(float v, float* sh) {
#pragma unroll
  for (int o = 16; o > 0; o >>= 1) v = fmaxf(v, __shfl_xor(v, o, 32));
  const int w  = threadIdx.x >> 5;
  const int nw = blockDim.x >> 5;
  __syncthreads();
  if ((threadIdx.x & 31) == 0) sh[w] = v;
  __syncthreads();
  if (threadIdx.x < 32) {
    float t = (threadIdx.x < nw) ? sh[threadIdx.x] : -INFINITY;
#pragma unroll
    for (int o = 16; o > 0; o >>= 1) t = fmaxf(t, __shfl_xor(t, o, 32));
    if (threadIdx.x == 0) sh[0] = t;
  }
  __syncthreads();
  return sh[0];
}

// ---------------- tiny setup kernels ----------------
__global__ void st_init_kernel(float* scal) { scal[0] = 0.0f; }

__global__ __launch_bounds__(256) void st_rownorm_kernel(
    const float* __restrict__ X, const int* __restrict__ idx, float* __restrict__ scal) {
  __shared__ float sh[64];
  const size_t row = (size_t)idx[blockIdx.x];
  const float* xr = X + row * NN;
  float ss = 0.0f;
#pragma unroll
  for (int i = 0; i < NN / 256; ++i) { float v = xr[threadIdx.x + (i << 8)]; ss += v * v; }
  ss = blockReduceSum(ss, sh);
  if (threadIdx.x == 0) atomicAdd(&scal[0], sqrtf(ss));
}

__global__ void st_finalize_kernel(float* scal) {
  // C = (sum_norm / m) / sqrt(n)  ->  invC = m*sqrt(n)/sum_norm
  scal[1] = ((float)MM * sqrtf((float)NN)) / scal[0];
}

__global__ __launch_bounds__(256) void st_scale_x_kernel(
    const float* __restrict__ x, const float* __restrict__ scal,
    float* __restrict__ xsOut, _Float16* __restrict__ xsH) {
  const float invC = scal[1];
  const size_t i = ((size_t)blockIdx.x * 256 + threadIdx.x) * 4;
  float4v v = *(const float4v*)(x + i);
  float4v o; half4 h;
#pragma unroll
  for (int j = 0; j < 4; ++j) { o[j] = v[j] * invC; h[j] = (_Float16)o[j]; }
  *(float4v*)(xsOut + i) = o;
  *(half4*)(xsH + i) = h;
}

__global__ __launch_bounds__(256) void st_gather_scale_kernel(
    const float* __restrict__ X, const int* __restrict__ idx,
    const float* __restrict__ scal, _Float16* __restrict__ XcH) {
  const float invC = scal[1];
  const size_t src = (size_t)idx[blockIdx.x] * NN;
  const size_t dst = (size_t)blockIdx.x * NN;
#pragma unroll
  for (int i = 0; i < NN / 256; ++i) {
    const int c = threadIdx.x + (i << 8);
    XcH[dst + c] = (_Float16)(X[src + c] * invC);
  }
}

__global__ __launch_bounds__(256) void st_cvt_f16_kernel(
    const float* __restrict__ src, _Float16* __restrict__ dst) {
  const size_t i = ((size_t)blockIdx.x * 256 + threadIdx.x) * 4;
  float4v v = *(const float4v*)(src + i);
  half4 h;
#pragma unroll
  for (int j = 0; j < 4; ++j) h[j] = (_Float16)v[j];
  *(half4*)(dst + i) = h;
}

// ---------------- WMMA GEMM tiling ----------------
// Block tile 128(M) x 128(N), K-step 32, 256 threads = 8 waves in a 4x2 grid:
// wave (wm in 0..3 -> 32-row strip, wn in 0..1 -> 64-col strip), each wave holds
// a 32x64 strip = 2x4 v_wmma_f32_16x16x32_f16 accumulators (8 WMMA / K-step,
// 12 ds_load_b128 / K-step -> 1.5 LDS reads per WMMA).
// LDS tiles stored [row][K] (K contiguous, +8 halves pad), matching CDNA5 WMMA
// 16-bit operand layouts:
//   A (16x32): lane l holds M=l&15; halves [h*8..h*8+7] and [16+h*8..], h=l>>4
//   B (32x16): lane l holds N=l&15; halves [h*8..h*8+7] and [16+h*8..], h=l>>4
//   C/D (16x16 f32): vgpr r -> (M=r+8*h, N=l&15)
constexpr int BLK_M = 128, BLK_N = 128, BLK_K = 32;
constexpr int LD_SH = BLK_K + 8;  // 40 halves = 80B rows; chunks stay 16B aligned

// D = A[M,K] * B[N,K]^T * scale, A/B f16 row-major, async double-buffered staging.
__global__ __launch_bounds__(256) void st_gemm_async_kernel(
    const _Float16* __restrict__ A, const _Float16* __restrict__ B,
    float* __restrict__ Out, int M, int N, int K, float scale) {
  __shared__ _Float16 shA[2][BLK_M * LD_SH];
  __shared__ _Float16 shB[2][BLK_N * LD_SH];
  const int tid  = threadIdx.x;
  const int lane = tid & 31;
  const int wv   = tid >> 5;
  const int wm   = wv & 3;    // 0..3 -> M offset wm*32
  const int wn   = wv >> 2;   // 0..1 -> N offset wn*64
  const int h    = lane >> 4; // lane half
  const int l15  = lane & 15;
  const int mBlock = blockIdx.y * BLK_M;
  const int nBlock = blockIdx.x * BLK_N;

  // Staging assignment: 512 16B chunks per operand tile, 2 per thread per operand.
  const int rowS = tid >> 2;          // 0..63 (second chunk at rowS+64)
  const int chS  = tid & 3;           // 0..3
  const uint32_t aLds0 = (uint32_t)(size_t)&shA[0][0] + (uint32_t)(rowS * LD_SH * 2 + chS * 16);
  const uint32_t bLds0 = (uint32_t)(size_t)&shB[0][0] + (uint32_t)(rowS * LD_SH * 2 + chS * 16);
  const uint32_t rowHop = (uint32_t)(64 * LD_SH * 2);
  const uint32_t bufA = (uint32_t)(BLK_M * LD_SH * 2);
  const uint32_t bufB = (uint32_t)(BLK_N * LD_SH * 2);
  const _Float16* aG = A + (size_t)(mBlock + rowS) * K + chS * 8;
  const _Float16* bG = B + (size_t)(nBlock + rowS) * K + chS * 8;

  float8 acc[2][4] = {};

  auto issueTile = [&](int buf, int k0) {
    const uint32_t ab = aLds0 + (uint32_t)buf * bufA;
    const uint32_t bb = bLds0 + (uint32_t)buf * bufB;
    async_copy_b128(ab,          aG + k0);
    async_copy_b128(ab + rowHop, aG + (size_t)64 * K + k0);
    async_copy_b128(bb,          bG + k0);
    async_copy_b128(bb + rowHop, bG + (size_t)64 * K + k0);
  };

  issueTile(0, 0);
  int buf = 0;
  for (int k0 = 0; k0 < K; k0 += BLK_K) {
    wait_async_all();
    __syncthreads();
    if (k0 + BLK_K < K) issueTile(buf ^ 1, k0 + BLK_K);

    // compute from current buffer
    half16 aop[2];
#pragma unroll
    for (int ms = 0; ms < 2; ++ms) {
      const int aRow = wm * 32 + ms * 16 + l15;
      const half8 alo = *(const half8*)(&shA[buf][aRow * LD_SH + h * 8]);
      const half8 ahi = *(const half8*)(&shA[buf][aRow * LD_SH + 16 + h * 8]);
      aop[ms] = __builtin_shufflevector(alo, ahi,
          0, 1, 2, 3, 4, 5, 6, 7, 8, 9, 10, 11, 12, 13, 14, 15);
    }
#pragma unroll
    for (int ns = 0; ns < 4; ++ns) {
      const int bRow = wn * 64 + ns * 16 + l15;
      const half8 blo = *(const half8*)(&shB[buf][bRow * LD_SH + h * 8]);
      const half8 bhi = *(const half8*)(&shB[buf][bRow * LD_SH + 16 + h * 8]);
      const half16 bop = __builtin_shufflevector(blo, bhi,
          0, 1, 2, 3, 4, 5, 6, 7, 8, 9, 10, 11, 12, 13, 14, 15);
#pragma unroll
      for (int ms = 0; ms < 2; ++ms) {
        acc[ms][ns] = __builtin_amdgcn_wmma_f32_16x16x32_f16(
            false, aop[ms], false, bop, (short)0, acc[ms][ns], false, false);
      }
    }
    buf ^= 1;
  }

#pragma unroll
  for (int ms = 0; ms < 2; ++ms) {
#pragma unroll
    for (int ns = 0; ns < 4; ++ns) {
      const int col = nBlock + wn * 64 + ns * 16 + l15;
#pragma unroll
      for (int r = 0; r < 8; ++r) {
        const int row = mBlock + wm * 32 + ms * 16 + h * 8 + r;
        Out[(size_t)row * N + col] = acc[ms][ns][r] * scale;
      }
    }
  }
}

// x_hat GEMM: A is f32 (attn, converted on stage), B is v_h [K, Bld] row-major
// (transposed while staging into shB[n][k]). Synchronous staging, same tiling.
__global__ __launch_bounds__(256) void st_gemm_xhat_kernel(
    const float* __restrict__ A, const _Float16* __restrict__ B,
    float* __restrict__ Out, int M, int N, int K, int Bld, float scale) {
  __shared__ _Float16 shA[BLK_M * LD_SH];
  __shared__ _Float16 shB[BLK_N * LD_SH];
  const int tid  = threadIdx.x;
  const int lane = tid & 31;
  const int wv   = tid >> 5;
  const int wm   = wv & 3;
  const int wn   = wv >> 2;
  const int h    = lane >> 4;
  const int l15  = lane & 15;
  const int mBlock = blockIdx.y * BLK_M;
  const int nBlock = blockIdx.x * BLK_N;

  float8 acc[2][4] = {};

  for (int k0 = 0; k0 < K; k0 += BLK_K) {
    __syncthreads();
    // A tile: 128 rows x 32 halves, f32 -> f16 (2 chunks/thread)
#pragma unroll
    for (int c = tid; c < BLK_M * 4; c += 256) {
      const int row = c >> 2, ch = c & 3;
      const float4v f0 = *(const float4v*)(A + (size_t)(mBlock + row) * K + k0 + ch * 8);
      const float4v f1 = *(const float4v*)(A + (size_t)(mBlock + row) * K + k0 + ch * 8 + 4);
      half8 hh;
#pragma unroll
      for (int i = 0; i < 4; ++i) { hh[i] = (_Float16)f0[i]; hh[i + 4] = (_Float16)f1[i]; }
      *(half8*)(&shA[row * LD_SH + ch * 8]) = hh;
    }
    // B tile: transpose v_h[k0..k0+31][nBlock..nBlock+127] into shB[n][k]
    {
      const int kr = tid >> 3, nch = tid & 7;
      const half8 v0 = *(const half8*)(B + (size_t)(k0 + kr) * Bld + nBlock + nch * 16);
      const half8 v1 = *(const half8*)(B + (size_t)(k0 + kr) * Bld + nBlock + nch * 16 + 8);
#pragma unroll
      for (int i = 0; i < 8; ++i) {
        shB[(nch * 16 + i) * LD_SH + kr]     = v0[i];
        shB[(nch * 16 + 8 + i) * LD_SH + kr] = v1[i];
      }
    }
    __syncthreads();

    half16 aop[2];
#pragma unroll
    for (int ms = 0; ms < 2; ++ms) {
      const int aRow = wm * 32 + ms * 16 + l15;
      const half8 alo = *(const half8*)(&shA[aRow * LD_SH + h * 8]);
      const half8 ahi = *(const half8*)(&shA[aRow * LD_SH + 16 + h * 8]);
      aop[ms] = __builtin_shufflevector(alo, ahi,
          0, 1, 2, 3, 4, 5, 6, 7, 8, 9, 10, 11, 12, 13, 14, 15);
    }
#pragma unroll
    for (int ns = 0; ns < 4; ++ns) {
      const int bRow = wn * 64 + ns * 16 + l15;
      const half8 blo = *(const half8*)(&shB[bRow * LD_SH + h * 8]);
      const half8 bhi = *(const half8*)(&shB[bRow * LD_SH + 16 + h * 8]);
      const half16 bop = __builtin_shufflevector(blo, bhi,
          0, 1, 2, 3, 4, 5, 6, 7, 8, 9, 10, 11, 12, 13, 14, 15);
#pragma unroll
      for (int ms = 0; ms < 2; ++ms) {
        acc[ms][ns] = __builtin_amdgcn_wmma_f32_16x16x32_f16(
            false, aop[ms], false, bop, (short)0, acc[ms][ns], false, false);
      }
    }
  }

#pragma unroll
  for (int ms = 0; ms < 2; ++ms) {
#pragma unroll
    for (int ns = 0; ns < 4; ++ns) {
      const int col = nBlock + wn * 64 + ns * 16 + l15;
#pragma unroll
      for (int r = 0; r < 8; ++r) {
        const int row = mBlock + wm * 32 + ms * 16 + h * 8 + r;
        Out[(size_t)row * N + col] = acc[ms][ns][r] * scale;
      }
    }
  }
}

// ---------------- LayerNorm over last dim P (P = 256 or 1024) ----------------
__global__ __launch_bounds__(256) void st_layernorm_kernel(
    const float* __restrict__ in, const float* __restrict__ g, const float* __restrict__ bb,
    float* __restrict__ outF32, _Float16* __restrict__ outF16, int P) {
  __shared__ float sh[64];
  const size_t base = (size_t)blockIdx.x * P;
  float vals[4];
  const int cnt = P >> 8;
  float s = 0.0f, ss = 0.0f;
  for (int i = 0; i < cnt; ++i) {
    const float v = in[base + threadIdx.x + (i << 8)];
    vals[i] = v; s += v; ss += v * v;
  }
  s  = blockReduceSum(s, sh);
  ss = blockReduceSum(ss, sh);
  const float mean = s / (float)P;
  const float var  = ss / (float)P - mean * mean;
  const float inv  = rsqrtf(var + 1e-5f);
  for (int i = 0; i < cnt; ++i) {
    const int c = threadIdx.x + (i << 8);
    const float y = (vals[i] - mean) * inv * g[c] + bb[c];
    if (outF32) outF32[base + c] = y;
    outF16[base + c] = (_Float16)y;
  }
}

// ---------------- softmax in-place over P columns ----------------
__global__ __launch_bounds__(256) void st_softmax_kernel(float* __restrict__ attn, int P) {
  __shared__ float sh[64];
  const size_t base = (size_t)blockIdx.x * P;
  float mx = -INFINITY;
  for (int c = threadIdx.x; c < P; c += 256) mx = fmaxf(mx, attn[base + c]);
  mx = blockReduceMax(mx, sh);
  float s = 0.0f;
  for (int c = threadIdx.x; c < P; c += 256) s += __expf(attn[base + c] - mx);
  s = blockReduceSum(s, sh);
  const float inv = 1.0f / s;
  for (int c = threadIdx.x; c < P; c += 256) attn[base + c] = __expf(attn[base + c] - mx) * inv;
}

// ---------------- host-side launch ----------------
extern "C" void kernel_launch(void* const* d_in, const int* in_sizes, int n_in,
                              void* d_out, int out_size, void* d_ws, size_t ws_size,
                              hipStream_t stream) {
  const float* x     = (const float*)d_in[0];
  const float* Xdata = (const float*)d_in[1];
  const int*   midx  = (const int*)d_in[2];
  const float* Wq    = (const float*)d_in[3];
  const float* Wk    = (const float*)d_in[4];
  const float* Wv    = (const float*)d_in[5];
  const float* gq    = (const float*)d_in[6];
  const float* bq    = (const float*)d_in[7];
  const float* gk    = (const float*)d_in[8];
  const float* bk    = (const float*)d_in[9];
  const float* gv    = (const float*)d_in[10];
  const float* bv    = (const float*)d_in[11];

  float* out      = (float*)d_out;
  float* out_xs   = out;                                   // [B, n]
  float* out_xhat = out + (size_t)BB * NN;                 // [B, n]
  float* out_attn = out + 2 * (size_t)BB * NN;             // [B, m]
  float* out_v    = out_attn + (size_t)BB * MM;            // [m, n]

  char* ws = (char*)d_ws;
  size_t cur = 0;
  auto alloc = [&](size_t bytes) -> void* {
    void* p = ws + cur;
    cur += (bytes + 255) & ~(size_t)255;
    return p;
  };
  float*    scal = (float*)alloc(2 * sizeof(float));
  _Float16* xs_h = (_Float16*)alloc((size_t)BB * NN * 2);
  _Float16* Xc_h = (_Float16*)alloc((size_t)MM * NN * 2);
  _Float16* Wq_h = (_Float16*)alloc((size_t)AA * NN * 2);
  _Float16* Wk_h = (_Float16*)alloc((size_t)AA * NN * 2);
  _Float16* Wv_h = (_Float16*)alloc((size_t)NN * NN * 2);
  _Float16* q_h  = (_Float16*)alloc((size_t)BB * AA * 2);
  _Float16* k_h  = (_Float16*)alloc((size_t)MM * AA * 2);
  _Float16* v_h  = (_Float16*)alloc((size_t)MM * NN * 2);
  float*    q_raw = (float*)alloc((size_t)BB * AA * 4);
  float*    k_raw = (float*)alloc((size_t)MM * AA * 4);
  (void)in_sizes; (void)n_in; (void)out_size; (void)ws_size;

  // 1) scalar normalizer C -> invC
  st_init_kernel<<<1, 1, 0, stream>>>(scal);
  st_rownorm_kernel<<<MM, 256, 0, stream>>>(Xdata, midx, scal);
  st_finalize_kernel<<<1, 1, 0, stream>>>(scal);

  // 2) xs (f32 out + f16 ws), gathered+scaled X_cross (f16 ws), weights -> f16
  st_scale_x_kernel<<<(BB * NN) / 1024, 256, 0, stream>>>(x, scal, out_xs, xs_h);
  st_gather_scale_kernel<<<MM, 256, 0, stream>>>(Xdata, midx, scal, Xc_h);
  st_cvt_f16_kernel<<<(AA * NN) / 1024, 256, 0, stream>>>(Wq, Wq_h);
  st_cvt_f16_kernel<<<(AA * NN) / 1024, 256, 0, stream>>>(Wk, Wk_h);
  st_cvt_f16_kernel<<<(NN * NN) / 1024, 256, 0, stream>>>(Wv, Wv_h);

  // 3) QKV projections (WMMA f32-accumulate, async double-buffered staging)
  st_gemm_async_kernel<<<dim3(AA / BLK_N, BB / BLK_M), 256, 0, stream>>>(
      xs_h, Wq_h, q_raw, BB, AA, NN, 1.0f);
  st_gemm_async_kernel<<<dim3(AA / BLK_N, MM / BLK_M), 256, 0, stream>>>(
      Xc_h, Wk_h, k_raw, MM, AA, NN, 1.0f);
  st_gemm_async_kernel<<<dim3(NN / BLK_N, MM / BLK_M), 256, 0, stream>>>(
      Xc_h, Wv_h, out_v, MM, NN, NN, 1.0f);

  // 4) per-stream LayerNorm (v normalized in place into d_out, plus f16 copy)
  st_layernorm_kernel<<<BB, 256, 0, stream>>>(q_raw, gq, bq, nullptr, q_h, AA);
  st_layernorm_kernel<<<MM, 256, 0, stream>>>(k_raw, gk, bk, nullptr, k_h, AA);
  st_layernorm_kernel<<<MM, 256, 0, stream>>>(out_v, gv, bv, out_v, v_h, NN);

  // 5) attention scores (scaled), softmax in place
  st_gemm_async_kernel<<<dim3(MM / BLK_N, BB / BLK_M), 256, 0, stream>>>(
      q_h, k_h, out_attn, BB, MM, AA, 0.0625f /* 1/sqrt(256) */);
  st_softmax_kernel<<<BB, 256, 0, stream>>>(out_attn, MM);

  // 6) x_hat = attn @ v
  st_gemm_xhat_kernel<<<dim3(NN / BLK_N, BB / BLK_M), 256, 0, stream>>>(
      out_attn, v_h, out_xhat, BB, NN, MM, NN, 1.0f);
}
